// ResidualAxialBlock_90855738179586
// MI455X (gfx1250) — matrix operand order repaired and checked by reading, not compiled, not run on hardware
//
#include <hip/hip_runtime.h>
#include <hip/hip_bf16.h>

typedef __attribute__((ext_vector_type(16))) __bf16 v16bf;
typedef __attribute__((ext_vector_type(8)))  __bf16 v8bf;
typedef __attribute__((ext_vector_type(8)))  float  v8f;
typedef __attribute__((ext_vector_type(4)))  float  v4f;
typedef __attribute__((ext_vector_type(2)))  float  v2f;

// round-to-nearest-even f32 -> bf16 bits
__device__ __forceinline__ unsigned short f2bf(float f) {
  unsigned int u = __float_as_uint(f);
  u += 0x7FFFu + ((u >> 16) & 1u);
  return (unsigned short)(u >> 16);
}

__device__ __forceinline__ unsigned int f2bf_pk(float a, float b) {
  return (unsigned int)f2bf(a) | ((unsigned int)f2bf(b) << 16);
}

// A fragment, 16-bit elements, 16xK tile at k-step kk
// lanes 0-15 hold K = {0..7, 16..23}; lanes 16-31 hold K = {8..15, 24..31} (rel. to kk*32)
__device__ __forceinline__ v16bf load_a16(const unsigned short* row, int kk, int lane) {
  int base = kk * 32 + ((lane & 16) ? 8 : 0);
  union { v16bf v; v8bf h[2]; } u;
  u.h[0] = *(const v8bf*)(row + base);
  u.h[1] = *(const v8bf*)(row + base + 16);
  return u.v;
}

// B fragment, 16-bit: lane n holds 16 K of column n
// lanes 0-15: K = {0..7, 16..23}; lanes 16-31: K = {8..15, 24..31} (rel. to kk*32); row = K-major transposed weight
__device__ __forceinline__ v16bf load_b16(const unsigned short* row, int kk, int lane) {
  int base = kk * 32 + ((lane & 16) ? 8 : 0);
  union { v16bf v; v8bf h[2]; } u;
  u.h[0] = *(const v8bf*)(row + base);
  u.h[1] = *(const v8bf*)(row + base + 16);
  return u.v;
}

__device__ __forceinline__ v4f shfl_xor_v4(v4f v, int mask) {
  v4f r;
  r[0] = __shfl_xor(v[0], mask, 32);
  r[1] = __shfl_xor(v[1], mask, 32);
  r[2] = __shfl_xor(v[2], mask, 32);
  r[3] = __shfl_xor(v[3], mask, 32);
  return r;
}

// ---------------------------------------------------------------------------
// Prep: transpose + bf16-convert weights for one axis into workspace:
//   [0,65536)        WqT  [128][512]
//   [65536,131072)   WkT  [128][512]
//   [131072,196608)  WvT  [128][512]
//   [196608,262144)  WoT  [512][128]
// ---------------------------------------------------------------------------
__global__ __launch_bounds__(256) void prep_weights(
    const float* __restrict__ Wq, const float* __restrict__ Wkv,
    const float* __restrict__ Wo, unsigned short* __restrict__ dst)
{
  int i = blockIdx.x * 256 + threadIdx.x;  // 0..262143
  unsigned short v;
  if (i < 65536)       { int n = i >> 9, c = i & 511;                     v = f2bf(Wq[c * 128 + n]); }
  else if (i < 131072) { int j = i - 65536;  int n = j >> 9, c = j & 511; v = f2bf(Wkv[c * 256 + n]); }
  else if (i < 196608) { int j = i - 131072; int n = j >> 9, c = j & 511; v = f2bf(Wkv[c * 256 + 128 + n]); }
  else                 { int j = i - 196608; int n = j >> 7, k = j & 127; v = f2bf(Wo[k * 512 + n]); }
  dst[i] = v;
}

// ---------------------------------------------------------------------------
// Fused axial attention for one axis. One block = one length-64 sequence.
// LDS layout (160 KB dynamic):
//   [0,64K)    Xs  bf16 [64][512]   (phase 0/1); phase 2 reuses +32K as:
//              Os  f32 [64][128] at +32KB
//   [64K,96K)  Qs  f32 [64][128]    -- reused in phase 3 as Ob bf16 [64][128]
//   [96K,128K) Ks  f32 [64][128]
//   [128K,160K)Vs  f32 [64][128]    (channel-rotated: V[j][c] at j*128 + ((c+j*4)&127))
// ---------------------------------------------------------------------------
__global__ __launch_bounds__(256) void axial_attn_kernel(
    const float* __restrict__ x, const float* __restrict__ src,
    const unsigned short* __restrict__ wT, const float* __restrict__ bo,
    float* __restrict__ out, int axis)
{
  extern __shared__ char smem[];
  unsigned short* Xs = (unsigned short*)smem;
  float* Os = (float*)(smem + 32768);
  float* Qs = (float*)(smem + 65536);
  float* Ks = (float*)(smem + 98304);
  float* Vs = (float*)(smem + 131072);
  unsigned short* Ob = (unsigned short*)(smem + 65536);

  const int tid  = threadIdx.x;
  const int lane = tid & 31;
  const int wave = tid >> 5;
  const int l15  = lane & 15;
  const int half = (lane >> 4) & 1;

  const int blk = blockIdx.x;
  int tokStride, seqBase;
  if (axis == 0) { tokStride = 64 * 512; seqBase = (blk >> 6) * (64 * 64 * 512) + (blk & 63) * 512; }
  else           { tokStride = 512;      seqBase = blk * (64 * 512); }

  // ---- Phase 0: load X sequence -> bf16 LDS (coalesced float4, packed b64 stores)
  const float* xb = x + seqBase;
  #pragma unroll
  for (int it = 0; it < 32; ++it) {
    int f   = tid + it * 256;        // float4 index 0..8191
    int tok = f >> 7;
    int pos = (f & 127) * 4;
    const float4 v = *(const float4*)(xb + tok * tokStride + pos);
    uint2 p; p.x = f2bf_pk(v.x, v.y); p.y = f2bf_pk(v.z, v.w);
    *(uint2*)(Xs + tok * 512 + pos) = p;
  }
  __syncthreads();

  // ---- Phase 1: Q,K,V = X @ W   (bf16 WMMA, K=512 -> 16 k-steps)
  for (int t = wave; t < 96; t += 8) {       // 3 mats * 8 ntiles * 4 mtiles
    int mat = t / 32;
    int rem = t - mat * 32;
    int nt = rem >> 2, mt = rem & 3;
    const unsigned short* arow = Xs + (mt * 16 + l15) * 512;
    const unsigned short* brow = wT + mat * 65536 + (nt * 16 + l15) * 512;
    v8f acc = {0.f,0.f,0.f,0.f,0.f,0.f,0.f,0.f};
    #pragma unroll
    for (int kk = 0; kk < 16; ++kk) {
      v16bf a = load_a16(arow, kk, lane);
      v16bf b = load_b16(brow, kk, lane);
      acc = __builtin_amdgcn_wmma_f32_16x16x32_bf16(false, a, false, b, (short)0, acc, false, false);
    }
    if (mat == 2) {
      // V: channel-rotated layout for conflict-free float4 reads in phase 2
      #pragma unroll
      for (int r = 0; r < 8; ++r) {
        int j = mt * 16 + half * 8 + r;       // token
        int c = nt * 16 + l15;                // channel
        Vs[j * 128 + ((c + j * 4) & 127)] = acc[r];
      }
    } else {
      float* dst = (mat == 0) ? Qs : Ks;
      float scale = (mat == 0) ? 0.5f : 1.0f;  // dim_heads^-0.5 folded into Q
      #pragma unroll
      for (int r = 0; r < 8; ++r)
        dst[(mt * 16 + r + half * 8) * 128 + nt * 16 + l15] = acc[r] * scale;
    }
  }
  __syncthreads();

  // ---- Phase 2: attention core. 128 items = (32 heads x 4 q-strips), 16 per wave.
  for (int q = 0; q < 16; ++q) {
    int it = wave * 16 + q;
    int h = it >> 2, mt = it & 3;

    // dots row-strip [16 x 64] via f32 WMMA 16x16x4 (K = dim_heads = 4)
    v2f a;
    { const float* qr = Qs + (mt * 16 + l15) * 128 + h * 4 + half * 2;
      a.x = qr[0]; a.y = qr[1]; }
    v8f tjs[4];
    #pragma unroll
    for (int jt = 0; jt < 4; ++jt) {
      v2f b;
      const float* kr = Ks + (jt * 16 + l15) * 128 + h * 4 + half * 2;
      b.x = kr[0]; b.y = kr[1];
      v8f z = {0.f,0.f,0.f,0.f,0.f,0.f,0.f,0.f};
      tjs[jt] = __builtin_amdgcn_wmma_f32_16x16x4_f32(false, a, false, b, (short)0, z, false, false);
    }

    // softmax across j, probs stay in registers
    #pragma unroll
    for (int r = 0; r < 8; ++r) {
      float m = fmaxf(fmaxf(tjs[0][r], tjs[1][r]), fmaxf(tjs[2][r], tjs[3][r]));
      #pragma unroll
      for (int msk = 1; msk <= 8; msk <<= 1) m = fmaxf(m, __shfl_xor(m, msk, 32));
      float s = 0.f;
      #pragma unroll
      for (int jt = 0; jt < 4; ++jt) { float e = __expf(tjs[jt][r] - m); tjs[jt][r] = e; s += e; }
      #pragma unroll
      for (int msk = 1; msk <= 8; msk <<= 1) s += __shfl_xor(s, msk, 32);
      float inv = 1.f / s;
      #pragma unroll
      for (int jt = 0; jt < 4; ++jt) tjs[jt][r] *= inv;
    }

    // o strip [16 x 4] = P @ V_h : packed v4f FMAs + 16-lane butterfly over j
    v4f vreg[4];
    #pragma unroll
    for (int jt = 0; jt < 4; ++jt) {
      int j = jt * 16 + l15;
      vreg[jt] = *(const v4f*)(Vs + j * 128 + ((h * 4 + j * 4) & 127));
    }
    #pragma unroll
    for (int r = 0; r < 8; ++r) {
      v4f part = vreg[0] * tjs[0][r] + vreg[1] * tjs[1][r]
               + vreg[2] * tjs[2][r] + vreg[3] * tjs[3][r];
      #pragma unroll
      for (int msk = 1; msk <= 8; msk <<= 1) part += shfl_xor_v4(part, msk);
      if (l15 == 0) {
        int row = mt * 16 + half * 8 + r;
        *(v4f*)(Os + row * 128 + h * 4) = part;   // disjoint per item
      }
    }
  }
  __syncthreads();

  // ---- Phase 3: out = src + bo + O @ Wo   (convert O to bf16, reuse Qs region)
  #pragma unroll
  for (int it = 0; it < 16; ++it) {
    int i = (tid + it * 256) * 2;   // float pair index
    *(unsigned int*)(Ob + i) = f2bf_pk(Os[i], Os[i + 1]);
  }
  __syncthreads();

  const unsigned short* WoT = wT + 196608;
  const float* srcb = src + seqBase;
  float* outb = out + seqBase;
  for (int t = wave; t < 128; t += 8) {     // 32 ntiles * 4 mtiles
    int nt = t >> 2, mt = t & 3;
    const unsigned short* arow = Ob + (mt * 16 + l15) * 128;
    const unsigned short* brow = WoT + (nt * 16 + l15) * 128;
    v8f acc = {0.f,0.f,0.f,0.f,0.f,0.f,0.f,0.f};
    #pragma unroll
    for (int kk = 0; kk < 4; ++kk) {
      v16bf a = load_a16(arow, kk, lane);
      v16bf b = load_b16(brow, kk, lane);
      acc = __builtin_amdgcn_wmma_f32_16x16x32_bf16(false, a, false, b, (short)0, acc, false, false);
    }
    int c = nt * 16 + l15;
    float bias = bo[c];
    #pragma unroll
    for (int r = 0; r < 8; ++r) {
      int i = mt * 16 + r + half * 8;
      int g = i * tokStride + c;
      outb[g] = srcb[g] + bias + acc[r];
    }
  }
}

extern "C" void kernel_launch(void* const* d_in, const int* in_sizes, int n_in,
                              void* d_out, int out_size, void* d_ws, size_t ws_size,
                              hipStream_t stream)
{
  const float* x     = (const float*)d_in[0];
  const float* Wq_h  = (const float*)d_in[1];
  const float* Wkv_h = (const float*)d_in[2];
  const float* Wo_h  = (const float*)d_in[3];
  const float* bo_h  = (const float*)d_in[4];
  const float* Wq_w  = (const float*)d_in[5];
  const float* Wkv_w = (const float*)d_in[6];
  const float* Wo_w  = (const float*)d_in[7];
  const float* bo_w  = (const float*)d_in[8];
  float* out = (float*)d_out;

  unsigned short* w0 = (unsigned short*)d_ws;   // 262144 halfs per axis (1 MB total)
  unsigned short* w1 = w0 + 262144;

  prep_weights<<<1024, 256, 0, stream>>>(Wq_h, Wkv_h, Wo_h, w0);
  prep_weights<<<1024, 256, 0, stream>>>(Wq_w, Wkv_w, Wo_w, w1);

  const size_t lds = 160 * 1024;
  // branch H: out = x + attn_h(x)   (each block owns x[b, :, w, :] exclusively)
  axial_attn_kernel<<<2048, 256, lds, stream>>>(x, x,   w0, bo_h, out, 0);
  // branch W: out += attn_w(x)      (each block owns x[b, h, :, :] exclusively)
  axial_attn_kernel<<<2048, 256, lds, stream>>>(x, out, w1, bo_w, out, 1);
}
